// StructureAutoencoder_25881472925792
// MI455X (gfx1250) — matrix operand order repaired and checked by reading, not compiled, not run on hardware
//
#include <hip/hip_runtime.h>

// ---------------------------------------------------------------------------
// Problem constants (from reference)
// ---------------------------------------------------------------------------
#define NRES   4096
#define KNB    48
#define PDIM   128
#define ATM    5
#define NBINS  16
#define RPC    32
#define FDIM   480          // fused feature dim: 400 dist + 16 dmap + 15 dir + 9 rot + 15 vec + pad
#define ROWS   64           // (n,k) rows per block
#define LN_EPS 1e-5f

typedef __attribute__((ext_vector_type(16))) _Float16 v16h;
typedef __attribute__((ext_vector_type(8)))  _Float16 v8h;
typedef __attribute__((ext_vector_type(8)))  float    v8f;

// ---------------------------------------------------------------------------
// A-fragment loader (16-bit A, 16x32, wave32):
//   lanes 0-15 : row M = lane,      K = 0..7  (v0-3) and 16..23 (v4-7)
//   lanes 16-31: row M = lane-16,   K = 8..15 (v0-3) and 24..31 (v4-7)
// base: row-major f16 in LDS, row stride `stride` halves.
// ---------------------------------------------------------------------------
__device__ __forceinline__ v16h load_a_frag(const _Float16* base, int stride,
                                            int mrow0, int kbase) {
  const int lane = threadIdx.x & 31;
  const int row  = mrow0 + (lane & 15);
  const int g    = lane >> 4;
  const _Float16* p = base + row * stride + kbase + g * 8;
  v8h lo = *(const v8h*)(p);        // K = g*8 + 0..7
  v8h hi = *(const v8h*)(p + 16);   // K = 16 + g*8 + 0..7
  v16h a;
#pragma unroll
  for (int i = 0; i < 8; ++i) { a[i] = lo[i]; a[i + 8] = hi[i]; }
  return a;
}

__device__ __forceinline__ float gelu_tanh(float x) {
  const float c = 0.7978845608028654f;   // sqrt(2/pi)
  float t = tanhf(c * (x + 0.044715f * x * x * x));
  return 0.5f * x * (1.0f + t);
}

// ---------------------------------------------------------------------------
// Frames: R (row-major 3x3, columns = e1,e2,e3) and ca per residue
// ---------------------------------------------------------------------------
__global__ void frames_kernel(const float* __restrict__ pos,
                              float* __restrict__ Rws, float* __restrict__ caws) {
  int n = blockIdx.x * blockDim.x + threadIdx.x;
  if (n >= NRES) return;
  const float* p = pos + (size_t)n * ATM * 3;
  float nx = p[0], ny = p[1], nz = p[2];
  float cax = p[3], cay = p[4], caz = p[5];
  float cx = p[6], cy = p[7], cz = p[8];

  float e1x = cx - cax, e1y = cy - cay, e1z = cz - caz;
  float inv = rsqrtf(fmaxf(e1x*e1x + e1y*e1y + e1z*e1z, 1e-6f));
  e1x *= inv; e1y *= inv; e1z *= inv;

  float v2x = nx - cax, v2y = ny - cay, v2z = nz - caz;
  float d = v2x*e1x + v2y*e1y + v2z*e1z;
  float e2x = v2x - d*e1x, e2y = v2y - d*e1y, e2z = v2z - d*e1z;
  inv = rsqrtf(fmaxf(e2x*e2x + e2y*e2y + e2z*e2z, 1e-6f));
  e2x *= inv; e2y *= inv; e2z *= inv;

  float e3x = e1y*e2z - e1z*e2y;
  float e3y = e1z*e2x - e1x*e2z;
  float e3z = e1x*e2y - e1y*e2x;

  float* R = Rws + n * 9;       // R[i*3 + col], columns are e1,e2,e3
  R[0] = e1x; R[1] = e2x; R[2] = e3x;
  R[3] = e1y; R[4] = e2y; R[5] = e3y;
  R[6] = e1z; R[7] = e2z; R[8] = e3z;
  caws[n*3+0] = cax; caws[n*3+1] = cay; caws[n*3+2] = caz;
}

// ---------------------------------------------------------------------------
// Pack fused feature weights [FDIM x 128] fp32 concat -> f16 B-fragment order:
//   idx = (((kt*8 + nt)*32 + lane)*16 + j)
//   col = nt*16 + (lane&15),  kk = kt*32 + (lane>>4)*8 + (j&7) + (j>>3)*16
// ---------------------------------------------------------------------------
__global__ void pack_feat_w(const float* __restrict__ Wdist,
                            const float* __restrict__ Wdmap,
                            const float* __restrict__ Wdir,
                            const float* __restrict__ Wrot,
                            const float* __restrict__ Wvec,
                            _Float16* __restrict__ outp) {
  int idx = blockIdx.x * blockDim.x + threadIdx.x;
  const int total = (FDIM / 32) * 8 * 32 * 16;   // 61440
  if (idx >= total) return;
  int j    = idx & 15;
  int lane = (idx >> 4) & 31;
  int t    = idx >> 9;
  int nt   = t & 7;
  int kt   = t >> 3;
  int col  = nt * 16 + (lane & 15);
  int kk   = kt * 32 + (lane >> 4) * 8 + (j & 7) + (j >> 3) * 16;
  float v = 0.0f;
  if      (kk < 400) v = Wdist[kk * PDIM + col];
  else if (kk < 416) v = Wdmap[(kk - 400) * PDIM + col];
  else if (kk < 431) v = Wdir[(kk - 416) * PDIM + col];
  else if (kk < 440) v = Wrot[(kk - 431) * PDIM + col];
  else if (kk < 455) v = Wvec[(kk - 440) * PDIM + col];
  outp[idx] = (_Float16)v;
}

__global__ void pack_w(const float* __restrict__ W, _Float16* __restrict__ outp,
                       int Krows, int Ncols) {
  int NT = Ncols >> 4;
  int total = (Krows >> 5) * NT * 512;
  int idx = blockIdx.x * blockDim.x + threadIdx.x;
  if (idx >= total) return;
  int j    = idx & 15;
  int lane = (idx >> 4) & 31;
  int t    = idx >> 9;
  int nt   = t % NT;
  int kt   = t / NT;
  int col  = nt * 16 + (lane & 15);
  int kk   = kt * 32 + (lane >> 4) * 8 + (j & 7) + (j >> 3) * 16;
  outp[idx] = (_Float16)W[kk * Ncols + col];
}

// ---------------------------------------------------------------------------
// Fused main kernel: features -> GEMM (+relpos) -> LN -> MLP (WMMA x3)
// 256 threads = 8 wave32, 64 rows per block.
// ---------------------------------------------------------------------------
__global__ __launch_bounds__(256)
void fused_pair_kernel(const float* __restrict__ pos,
                       const float* __restrict__ dmap,
                       const float* __restrict__ mask,
                       const float* __restrict__ W_relpos,
                       const float* __restrict__ ln_scale,
                       const float* __restrict__ ln_offset,
                       const float* __restrict__ b1,
                       const float* __restrict__ b2,
                       const int*   __restrict__ neighbours,
                       const int*   __restrict__ resi,
                       const int*   __restrict__ chain,
                       const int*   __restrict__ batch,
                       const float* __restrict__ Rws,
                       const float* __restrict__ caws,
                       const _Float16* __restrict__ Wfeatp,
                       const _Float16* __restrict__ W1p,
                       const _Float16* __restrict__ W2p,
                       float* __restrict__ out_pair,
                       float* __restrict__ out_mask) {
  extern __shared__ char smem[];
  _Float16* feat    = (_Float16*)smem;                               // 64*480 f16 = 61440 B
  float*    pairbuf = (float*)(smem + ROWS * FDIM * 2);              // 64*128 f32 = 32768 B
  _Float16* act     = (_Float16*)((char*)pairbuf + ROWS * PDIM * 4); // 64*128 f16 = 16384 B
  _Float16* hbuf    = (_Float16*)((char*)act + ROWS * PDIM * 2);     // 64*256 f16 = 32768 B
  int*      aux_jj  = (int*)((char*)hbuf + ROWS * 256 * 2);          // 64 ints
  int*      aux_cls = aux_jj + ROWS;                                 // 64 ints

  const int tid  = threadIdx.x;
  const int lane = tid & 31;
  const int wave = tid >> 5;
  const int r0   = blockIdx.x * ROWS;
  const float step = 22.0f / (float)NBINS;

  // ---- Stage 1: per-row small features (64 active threads) ----
  if (tid < ROWS) {
    int r = r0 + tid;
    int n = r / KNB;
    int j = neighbours[r];
    int jj = (j < 0 || j >= NRES) ? 0 : j;
    float mval = mask[n] * mask[jj] * (j != -1 ? 1.0f : 0.0f);
    out_mask[r] = mval;
    aux_jj[tid] = jj;

    int rel = resi[jj] - resi[n];
    rel = min(max(rel, -RPC), RPC) + RPC;
    bool same = (chain[jj] == chain[n]) && (batch[jj] == batch[n]);
    aux_cls[tid] = same ? rel : (2 * RPC + 1);

    float rn[9], rj[9], can[3];
#pragma unroll
    for (int i = 0; i < 9; ++i) { rn[i] = Rws[n * 9 + i]; rj[i] = Rws[jj * 9 + i]; }
    can[0] = caws[n*3]; can[1] = caws[n*3+1]; can[2] = caws[n*3+2];

    _Float16* fr = feat + tid * FDIM;

    // direction (416..430) and pair-vector (440..454) features
#pragma unroll
    for (int a = 0; a < ATM; ++a) {
      const float* pj = pos + ((size_t)jj * ATM + a) * 3;
      float rx = pj[0] - can[0], ry = pj[1] - can[1], rz = pj[2] - can[2];
      float l0 = rn[0]*rx + rn[3]*ry + rn[6]*rz;   // R^T rel
      float l1 = rn[1]*rx + rn[4]*ry + rn[7]*rz;
      float l2 = rn[2]*rx + rn[5]*ry + rn[8]*rz;
      fr[440 + a*3 + 0] = (_Float16)l0;
      fr[440 + a*3 + 1] = (_Float16)l1;
      fr[440 + a*3 + 2] = (_Float16)l2;
      float inv = rsqrtf(fmaxf(l0*l0 + l1*l1 + l2*l2, 1e-6f));
      fr[416 + a*3 + 0] = (_Float16)(l0 * inv);
      fr[416 + a*3 + 1] = (_Float16)(l1 * inv);
      fr[416 + a*3 + 2] = (_Float16)(l2 * inv);
    }

    // rotation features (431..439): Rrel = R[n]^T R[j]
#pragma unroll
    for (int i = 0; i < 3; ++i)
#pragma unroll
      for (int l = 0; l < 3; ++l) {
        float s = rn[0*3+i]*rj[0*3+l] + rn[1*3+i]*rj[1*3+l] + rn[2*3+i]*rj[2*3+l];
        fr[431 + i*3 + l] = (_Float16)s;
      }

    // masked dmap RBF (400..415)
    float d = dmap[(size_t)n * NRES + jj];
#pragma unroll
    for (int b = 0; b < NBINS; ++b) {
      float u = (d - step * (float)b) / step;
      float val = (mval > 0.0f) ? __expf(-u * u) : 0.0f;
      fr[400 + b] = (_Float16)val;
    }

    // zero pad (455..479)
#pragma unroll
    for (int c = 455; c < FDIM; ++c) fr[c] = (_Float16)0.0f;
  }
  __syncthreads();

  // ---- Stage 2: all-atom distance RBF features (0..399) ----
  for (int idx = tid; idx < ROWS * 25; idx += 256) {
    int rl = idx / 25;
    int pr = idx % 25;
    int a1 = pr / 5, a2 = pr % 5;
    int r = r0 + rl;
    int n = r / KNB;
    int jj = aux_jj[rl];
    const float* p1 = pos + ((size_t)n  * ATM + a1) * 3;
    const float* p2 = pos + ((size_t)jj * ATM + a2) * 3;
    float dx = p1[0] - p2[0], dy = p1[1] - p2[1], dz = p1[2] - p2[2];
    float d = sqrtf(fmaxf(dx*dx + dy*dy + dz*dz, 1e-12f));
    _Float16* fr = feat + rl * FDIM + pr * NBINS;
#pragma unroll
    for (int b = 0; b < NBINS; ++b) {
      float u = (d - step * (float)b) / step;
      fr[b] = (_Float16)__expf(-u * u);
    }
  }
  __syncthreads();

  const int rlane = (lane >> 4) * 8;   // C-fragment row base within 16-tile
  const int clane = lane & 15;

  // ---- GEMM1: feat(64x480) x Wfeat(480x128) + relpos lookup -> pairbuf ----
  {
    const int mt  = wave >> 1;          // 0..3
    const int nt0 = (wave & 1) * 4;     // 0 or 4 (8 n-tiles)
    v8f acc[4] = {};
    for (int kt = 0; kt < FDIM / 32; ++kt) {
      v16h a = load_a_frag(feat, FDIM, mt * 16, kt * 32);
#pragma unroll
      for (int q = 0; q < 4; ++q) {
        v16h b = ((const v16h*)Wfeatp)[(kt * 8 + nt0 + q) * 32 + lane];
        acc[q] = __builtin_amdgcn_wmma_f32_16x16x32_f16(false, a, false, b,
                                                        (short)0, acc[q], false, false);
      }
    }
    int rbase = mt * 16 + rlane;
#pragma unroll
    for (int q = 0; q < 4; ++q) {
      int col = (nt0 + q) * 16 + clane;
#pragma unroll
      for (int v = 0; v < 8; ++v) {
        int rl = rbase + v;
        pairbuf[rl * PDIM + col] = acc[q][v] + W_relpos[aux_cls[rl] * PDIM + col];
      }
    }
  }
  __syncthreads();

  // ---- LayerNorm -> f16 activations ----
  if (tid < ROWS) {
    const float* pr = pairbuf + tid * PDIM;
    float mu = 0.0f;
#pragma unroll 8
    for (int c = 0; c < PDIM; ++c) mu += pr[c];
    mu *= (1.0f / PDIM);
    float var = 0.0f;
#pragma unroll 8
    for (int c = 0; c < PDIM; ++c) { float t = pr[c] - mu; var += t * t; }
    var *= (1.0f / PDIM);
    float inv = rsqrtf(var + LN_EPS);
#pragma unroll 8
    for (int c = 0; c < PDIM; ++c)
      act[tid * PDIM + c] = (_Float16)((pr[c] - mu) * inv * ln_scale[c] + ln_offset[c]);
  }
  __syncthreads();

  // ---- GEMM2: act(64x128) x W1(128x256) -> gelu -> hbuf ----
  {
    const int mt  = wave >> 1;
    const int nt0 = (wave & 1) * 8;     // 0 or 8 (16 n-tiles)
    v8f acc[8] = {};
    for (int kt = 0; kt < 4; ++kt) {
      v16h a = load_a_frag(act, PDIM, mt * 16, kt * 32);
#pragma unroll
      for (int q = 0; q < 8; ++q) {
        v16h b = ((const v16h*)W1p)[(kt * 16 + nt0 + q) * 32 + lane];
        acc[q] = __builtin_amdgcn_wmma_f32_16x16x32_f16(false, a, false, b,
                                                        (short)0, acc[q], false, false);
      }
    }
    int rbase = mt * 16 + rlane;
#pragma unroll
    for (int q = 0; q < 8; ++q) {
      int col = (nt0 + q) * 16 + clane;
      float bias = b1[col];
#pragma unroll
      for (int v = 0; v < 8; ++v)
        hbuf[(rbase + v) * 256 + col] = (_Float16)gelu_tanh(acc[q][v] + bias);
    }
  }
  __syncthreads();

  // ---- GEMM3: hbuf(64x256) x W2(256x128) + b2 -> global ----
  {
    const int mt  = wave >> 1;
    const int nt0 = (wave & 1) * 4;     // 0 or 4 (8 n-tiles)
    v8f acc[4] = {};
    for (int kt = 0; kt < 8; ++kt) {
      v16h a = load_a_frag(hbuf, 256, mt * 16, kt * 32);
#pragma unroll
      for (int q = 0; q < 4; ++q) {
        v16h b = ((const v16h*)W2p)[(kt * 8 + nt0 + q) * 32 + lane];
        acc[q] = __builtin_amdgcn_wmma_f32_16x16x32_f16(false, a, false, b,
                                                        (short)0, acc[q], false, false);
      }
    }
    int rbase = mt * 16 + rlane;
#pragma unroll
    for (int q = 0; q < 4; ++q) {
      int col = (nt0 + q) * 16 + clane;
      float bias = b2[col];
#pragma unroll
      for (int v = 0; v < 8; ++v) {
        int rl = rbase + v;
        out_pair[(size_t)(r0 + rl) * PDIM + col] = acc[q][v] + bias;
      }
    }
  }
}

// ---------------------------------------------------------------------------
// Launcher
// ---------------------------------------------------------------------------
extern "C" void kernel_launch(void* const* d_in, const int* in_sizes, int n_in,
                              void* d_out, int out_size, void* d_ws, size_t ws_size,
                              hipStream_t stream) {
  const float* pos       = (const float*)d_in[0];
  const float* dmap      = (const float*)d_in[1];
  const float* mask      = (const float*)d_in[2];
  const float* W_relpos  = (const float*)d_in[3];
  const float* W_dmap    = (const float*)d_in[4];
  const float* W_dist    = (const float*)d_in[5];
  const float* W_dir     = (const float*)d_in[6];
  const float* W_rot     = (const float*)d_in[7];
  const float* W_vec     = (const float*)d_in[8];
  const float* ln_scale  = (const float*)d_in[9];
  const float* ln_offset = (const float*)d_in[10];
  const float* W1        = (const float*)d_in[11];
  const float* b1        = (const float*)d_in[12];
  const float* W2        = (const float*)d_in[13];
  const float* b2        = (const float*)d_in[14];
  const int*   neighbours= (const int*)d_in[15];
  const int*   resi      = (const int*)d_in[16];
  const int*   chain     = (const int*)d_in[17];
  const int*   batch     = (const int*)d_in[18];

  // Workspace layout (256-byte aligned regions)
  char* ws = (char*)d_ws;
  float*     Rws    = (float*)(ws + 0);            // 4096*9*4   = 147456
  float*     caws   = (float*)(ws + 147456);       // 4096*3*4   = 49152
  _Float16*  Wfeatp = (_Float16*)(ws + 196608);    // 61440*2    = 122880
  _Float16*  W1p    = (_Float16*)(ws + 319488);    // 32768*2    = 65536
  _Float16*  W2p    = (_Float16*)(ws + 385024);    // 32768*2    = 65536

  float* out_pair = (float*)d_out;
  float* out_mask = out_pair + (size_t)NRES * KNB * PDIM;

  frames_kernel<<<(NRES + 255) / 256, 256, 0, stream>>>(pos, Rws, caws);
  pack_feat_w<<<(61440 + 255) / 256, 256, 0, stream>>>(W_dist, W_dmap, W_dir, W_rot, W_vec, Wfeatp);
  pack_w<<<(32768 + 255) / 256, 256, 0, stream>>>(W1, W1p, 128, 256);
  pack_w<<<(32768 + 255) / 256, 256, 0, stream>>>(W2, W2p, 256, 128);

  const size_t lds_bytes = (size_t)ROWS * FDIM * 2   // feat
                         + (size_t)ROWS * PDIM * 4   // pairbuf
                         + (size_t)ROWS * PDIM * 2   // act
                         + (size_t)ROWS * 256 * 2    // hbuf
                         + (size_t)ROWS * 2 * 4;     // aux
  const int nblocks = (NRES * KNB) / ROWS;           // 3072
  fused_pair_kernel<<<nblocks, 256, lds_bytes, stream>>>(
      pos, dmap, mask, W_relpos, ln_scale, ln_offset, b1, b2,
      neighbours, resi, chain, batch, Rws, caws, Wfeatp, W1p, W2p,
      out_pair, out_mask);
}
